// TransformerSimple_14989435863281
// MI455X (gfx1250) — matrix operand order repaired and checked by reading, not compiled, not run on hardware
//
#include <hip/hip_runtime.h>
#include <hip/hip_bf16.h>

typedef __attribute__((ext_vector_type(16))) _Float16 v16h;
typedef __attribute__((ext_vector_type(8)))  _Float16 h8v;
typedef __attribute__((ext_vector_type(8)))  float    v8f;
typedef __attribute__((ext_vector_type(4)))  int      v4i;

#define AS1 __attribute__((address_space(1)))
#define AS3 __attribute__((address_space(3)))

// ---------------------------------------------------------------------------
// CDNA5 async global->LDS copy (16B per lane), tracked by ASYNCcnt.
// Builtin signature (from hipcc diagnostic): (v4i AS1*, v4i AS3*, Ii, Ii).
// ---------------------------------------------------------------------------
__device__ __forceinline__ void async_g2l_16B(const void* g, void* l) {
#if __has_builtin(__builtin_amdgcn_global_load_async_to_lds_b128)
  __builtin_amdgcn_global_load_async_to_lds_b128((AS1 v4i*)(void*)g,
                                                 (AS3 v4i*)l, 0, 0);
#else
  unsigned loff = (unsigned)(unsigned long long)(AS3 void*)l;
  asm volatile("global_load_async_to_lds_b128 %0, %1, off"
               :: "v"(loff), "v"(g) : "memory");
#endif
}

__device__ __forceinline__ void wait_async0() {
#if __has_builtin(__builtin_amdgcn_s_wait_asynccnt)
  __builtin_amdgcn_s_wait_asynccnt(0);
#else
  asm volatile("s_wait_asynccnt 0x0" ::: "memory");
#endif
}

__device__ __forceinline__ void wait_async5() {
#if __has_builtin(__builtin_amdgcn_s_wait_asynccnt)
  __builtin_amdgcn_s_wait_asynccnt(5);
#else
  asm volatile("s_wait_asynccnt 0x5" ::: "memory");
#endif
}

// ---------------------------------------------------------------------------
// Fragment loaders per CDNA5 ISA VGPR layouts (CDNA5 ISA §7.12.2).
// A 16x32 f16: lane(half,m) -> halves [h*8..h*8+7] and [16+h*8..16+h*8+7].
// B 32x16 f16: lane column n=lane%16, halves [h*8..h*8+7] and [16+h*8..16+h*8+7].
// C/D 16x16 f32: reg v -> M = v + 8*half, N = lane%16.
// ---------------------------------------------------------------------------
__device__ __forceinline__ v16h load_a_frag(const _Float16* row, int half) {
  v16h r;
  const _Float16* p0 = row + half * 8;
  const _Float16* p1 = row + 16 + half * 8;
#pragma unroll
  for (int j = 0; j < 8; ++j) { r[j] = p0[j]; r[8 + j] = p1[j]; }
  return r;
}

__device__ __forceinline__ v16h load_b_frag(const _Float16* col, int half) {
  v16h r;
  const _Float16* p0 = col + half * 8;
  const _Float16* p1 = col + 16 + half * 8;
#pragma unroll
  for (int j = 0; j < 8; ++j) { r[j] = p0[j]; r[8 + j] = p1[j]; }
  return r;
}

// ---------------------------------------------------------------------------
// Embedding + sinusoidal PE: writes fp32 h (residual stream) + fp16 copy.
// ---------------------------------------------------------------------------
__global__ __launch_bounds__(256) void embed_pe_kernel(
    const int* __restrict__ x, const float* __restrict__ emb,
    float* __restrict__ H, _Float16* __restrict__ Hh) {
  int tok = blockIdx.x, d = threadIdx.x;
  int s = tok & 511;  // S = 512
  int id = x[tok];
  float e = emb[(size_t)id * 256 + d];
  int i2 = d & ~1;
  float ang = (float)s * __powf(10000.0f, -((float)i2) * (1.0f / 256.0f));
  float pe = (d & 1) ? __cosf(ang) : __sinf(ang);
  float y = e + pe;
  H[(size_t)tok * 256 + d] = y;
  Hh[(size_t)tok * 256 + d] = (_Float16)y;
}

// ---------------------------------------------------------------------------
// Weight fp32[K,N] -> fp16 transposed [N,K]  (makes B staging contiguous)
// ---------------------------------------------------------------------------
__global__ __launch_bounds__(256) void cvt_transpose_kernel(
    const float* __restrict__ W, _Float16* __restrict__ WT, int K, int N) {
  int idx = blockIdx.x * 256 + threadIdx.x;
  if (idx < K * N) {
    int kk = idx / N, n = idx % N;
    WT[(size_t)n * K + kk] = (_Float16)W[idx];
  }
}

// ---------------------------------------------------------------------------
// WMMA GEMM: C = act(A[M,K](f16) * B[K,N] + bias), B given transposed [N,K].
// Block = 256 threads (8 waves). Tile 256x64, K-step 32, double-buffered
// async global->LDS staging. Wave w: rows [w*32, w*32+32) -> 8 wmma/K-step.
// ---------------------------------------------------------------------------
#define GBM 256
#define GBN 64
#define GBK 32
#define LPAD 8

__global__ __launch_bounds__(256) void gemm_wmma_kernel(
    const _Float16* __restrict__ A, const _Float16* __restrict__ BT,
    const float* __restrict__ bias, float* __restrict__ C32,
    _Float16* __restrict__ C16, int M, int N, int K, int act) {
  __shared__ _Float16 As[2][GBM][GBK + LPAD];   // 2 x 256 x 40 halves
  __shared__ _Float16 Bs[2][GBN][GBK + LPAD];   // col-major: Bs[n][k]
  int t = threadIdx.x;
  int lane = t & 31, wave = t >> 5;
  int half = lane >> 4, ln = lane & 15;
  int m0 = blockIdx.y * GBM;
  int n0 = blockIdx.x * GBN;

  v8f acc[2][4] = {};

  // stage one K-tile: 4 async A chunks + 1 async B chunk per thread (5/wave)
  auto stage = [&](int buf, int k0) {
#pragma unroll
    for (int j = 0; j < 4; ++j) {
      int chunk = j * 256 + t;
      int r = chunk >> 2, c8 = (chunk & 3) * 8;
      async_g2l_16B(&A[(size_t)(m0 + r) * K + k0 + c8], &As[buf][r][c8]);
    }
    int n = t >> 2, k8 = (t & 3) * 8;
    async_g2l_16B(&BT[(size_t)(n0 + n) * K + k0 + k8], &Bs[buf][n][k8]);
  };

  int nk = K / GBK;
  stage(0, 0);
  for (int kt = 0; kt < nk; ++kt) {
    int buf = kt & 1;
    if (kt + 1 < nk) { stage(buf ^ 1, (kt + 1) * GBK); wait_async5(); }
    else             { wait_async0(); }
    __syncthreads();

    v16h bfr[4];
#pragma unroll
    for (int nt = 0; nt < 4; ++nt)
      bfr[nt] = load_b_frag(&Bs[buf][nt * 16 + ln][0], half);
#pragma unroll
    for (int mt = 0; mt < 2; ++mt) {
      v16h a = load_a_frag(&As[buf][wave * 32 + mt * 16 + ln][0], half);
#pragma unroll
      for (int nt = 0; nt < 4; ++nt)
        acc[mt][nt] = __builtin_amdgcn_wmma_f32_16x16x32_f16(
            false, a, false, bfr[nt], (short)0, acc[mt][nt], false, false);
    }
    __syncthreads();
  }

#pragma unroll
  for (int mt = 0; mt < 2; ++mt) {
    int mbase = m0 + wave * 32 + mt * 16 + half * 8;
#pragma unroll
    for (int nt = 0; nt < 4; ++nt) {
      int col = n0 + nt * 16 + ln;
      float bv = bias ? bias[col] : 0.0f;
#pragma unroll
      for (int v = 0; v < 8; ++v) {
        float xv = acc[mt][nt][v] + bv;
        if (act) xv = fmaxf(xv, 0.0f);
        size_t idx = (size_t)(mbase + v) * N + col;
        if (C32) C32[idx] = xv;
        if (C16) C16[idx] = (_Float16)xv;
      }
    }
  }
}

// ---------------------------------------------------------------------------
// Flash attention (causal), fp16 in/out. Grid: (B*H, S/128). 8 waves/block,
// wave w owns q rows [qbase+w*16, +16). 32-key K/V tiles in LDS: K staged
// async, V transposed manually. Online softmax via shfl_xor row reductions.
// ---------------------------------------------------------------------------
__global__ __launch_bounds__(256) void flash_attn_kernel(
    const _Float16* __restrict__ Q, const _Float16* __restrict__ Kb,
    const _Float16* __restrict__ V, _Float16* __restrict__ O) {
  __shared__ _Float16 Kt[32][64 + LPAD];      // [key][hd]
  __shared__ _Float16 Vt[64][32 + LPAD];      // [hd][key] transposed
  __shared__ _Float16 Pt[8][16][32 + LPAD];   // per-wave probabilities

  const int RS = 512;  // H*HD
  int t = threadIdx.x, lane = t & 31, wave = t >> 5;
  int half = lane >> 4, ln = lane & 15;
  int b = blockIdx.x >> 3, hh = blockIdx.x & 7;
  int qbase = blockIdx.y * 128;
  int qrow = qbase + wave * 16;

  const _Float16* qp = Q + (size_t)b * 512 * RS + hh * 64;
  const _Float16* kp = Kb + (size_t)b * 512 * RS + hh * 64;
  const _Float16* vp = V + (size_t)b * 512 * RS + hh * 64;

  // Q tile 16x64 as two A-frags: two contiguous 16B loads per lane per frag
  v16h aq[2];
  {
    const _Float16* qr = qp + (size_t)(qrow + ln) * RS;
#pragma unroll
    for (int s2 = 0; s2 < 2; ++s2) {
      h8v lo = *(const h8v*)(qr + s2 * 32 + half * 8);
      h8v hi = *(const h8v*)(qr + s2 * 32 + 16 + half * 8);
#pragma unroll
      for (int j = 0; j < 8; ++j) { aq[s2][j] = lo[j]; aq[s2][8 + j] = hi[j]; }
    }
  }

  v8f accO[4] = {};
  float mrow[8], lrow[8];
#pragma unroll
  for (int v = 0; v < 8; ++v) { mrow[v] = -3.0e38f; lrow[v] = 0.0f; }

  int nkt = (qbase + 128) >> 5;
  for (int kt = 0; kt < nkt; ++kt) {
    int kt0 = kt * 32;
    {
      int i = t >> 3, d8 = (t & 7) * 8;
      async_g2l_16B(&kp[(size_t)(kt0 + i) * RS + d8], &Kt[i][d8]);   // K: async
      h8v vv = *(const h8v*)&vp[(size_t)(kt0 + i) * RS + d8];        // V: transpose
#pragma unroll
      for (int j = 0; j < 8; ++j) Vt[d8 + j][i] = vv[j];
    }
    wait_async0();
    __syncthreads();

    if (kt0 <= qrow + 15) {
      v8f sc[2] = {};
#pragma unroll
      for (int nt = 0; nt < 2; ++nt) {
        const _Float16* kc = &Kt[nt * 16 + ln][0];
#pragma unroll
        for (int s2 = 0; s2 < 2; ++s2) {
          v16h bk;
#pragma unroll
          for (int j = 0; j < 16; ++j) bk[j] = kc[s2 * 32 + (j >> 3) * 16 + half * 8 + (j & 7)];
          sc[nt] = __builtin_amdgcn_wmma_f32_16x16x32_f16(
              false, aq[s2], false, bk, (short)0, sc[nt], false, false);
        }
      }
      float alpha[8], pb0[8], pb1[8];
      const float scale = 0.125f;  // 1/sqrt(64)
#pragma unroll
      for (int v = 0; v < 8; ++v) {
        int row = qrow + v + 8 * half;
        float s0 = sc[0][v] * scale, s1 = sc[1][v] * scale;
        if (kt0 + ln > row)      s0 = -3.0e38f;
        if (kt0 + 16 + ln > row) s1 = -3.0e38f;
        float mx = fmaxf(s0, s1);
#pragma unroll
        for (int off = 1; off < 16; off <<= 1) mx = fmaxf(mx, __shfl_xor(mx, off, 32));
        float mnew = fmaxf(mrow[v], mx);
        float p0 = __expf(s0 - mnew), p1 = __expf(s1 - mnew);
        float sum = p0 + p1;
#pragma unroll
        for (int off = 1; off < 16; off <<= 1) sum += __shfl_xor(sum, off, 32);
        alpha[v] = __expf(mrow[v] - mnew);
        lrow[v] = lrow[v] * alpha[v] + sum;
        mrow[v] = mnew;
        pb0[v] = p0; pb1[v] = p1;
      }
#pragma unroll
      for (int v = 0; v < 8; ++v) {
        Pt[wave][v + 8 * half][ln]      = (_Float16)pb0[v];
        Pt[wave][v + 8 * half][16 + ln] = (_Float16)pb1[v];
      }
      asm volatile("s_wait_dscnt 0x0" ::: "memory");  // wave-local visibility
#pragma unroll
      for (int nt = 0; nt < 4; ++nt)
#pragma unroll
        for (int v = 0; v < 8; ++v) accO[nt][v] = accO[nt][v] * alpha[v];
      v16h ap = load_a_frag(&Pt[wave][ln][0], half);
#pragma unroll
      for (int nt = 0; nt < 4; ++nt) {
        v16h bv = load_b_frag(&Vt[nt * 16 + ln][0], half);
        accO[nt] = __builtin_amdgcn_wmma_f32_16x16x32_f16(
            false, ap, false, bv, (short)0, accO[nt], false, false);
      }
    }
    __syncthreads();
  }

  _Float16* op = O + (size_t)b * 512 * RS + hh * 64;
#pragma unroll
  for (int v = 0; v < 8; ++v) {
    float inv = 1.0f / lrow[v];
    int row = qrow + v + 8 * half;
#pragma unroll
    for (int nt = 0; nt < 4; ++nt)
      op[(size_t)row * RS + nt * 16 + ln] = (_Float16)(accO[nt][v] * inv);
  }
}

// ---------------------------------------------------------------------------
// Y = LayerNorm(X + R); writes fp32 (residual stream) + fp16 (GEMM input).
// ---------------------------------------------------------------------------
__global__ __launch_bounds__(256) void residual_ln_kernel(
    const float* __restrict__ X, const float* __restrict__ R,
    const float* __restrict__ gs, const float* __restrict__ gb,
    float* __restrict__ Y, _Float16* __restrict__ Yh) {
  int tok = blockIdx.x, t = threadIdx.x;
  float v = X[(size_t)tok * 256 + t] + R[(size_t)tok * 256 + t];
  __shared__ float red[16];
  float s = v, s2 = v * v;
#pragma unroll
  for (int off = 16; off > 0; off >>= 1) {
    s  += __shfl_down(s, off, 32);
    s2 += __shfl_down(s2, off, 32);
  }
  int lane = t & 31, w = t >> 5;
  if (lane == 0) { red[w] = s; red[8 + w] = s2; }
  __syncthreads();
  if (t < 32) {
    float a  = (t < 8) ? red[t] : 0.0f;
    float b2 = (t < 8) ? red[8 + t] : 0.0f;
#pragma unroll
    for (int off = 4; off > 0; off >>= 1) {
      a  += __shfl_down(a, off, 32);
      b2 += __shfl_down(b2, off, 32);
    }
    if (t == 0) { red[0] = a; red[1] = b2; }
  }
  __syncthreads();
  float mean = red[0] * (1.0f / 256.0f);
  float var  = red[1] * (1.0f / 256.0f) - mean * mean;
  float y = (v - mean) * rsqrtf(var + 1e-5f) * gs[t] + gb[t];
  Y[(size_t)tok * 256 + t] = y;
  Yh[(size_t)tok * 256 + t] = (_Float16)y;
}

// ---------------------------------------------------------------------------
// Host orchestration
// ---------------------------------------------------------------------------
extern "C" void kernel_launch(void* const* d_in, const int* in_sizes, int n_in,
                              void* d_out, int out_size, void* d_ws, size_t ws_size,
                              hipStream_t stream) {
  (void)in_sizes; (void)n_in; (void)out_size; (void)ws_size;
  const int D = 256, Hh = 8, NL = 5, FF = 1024, OUT = 4096;
  const int B = 32, S = 512;
  const int BS = B * S;   // 16384 tokens
  const int DH = 512;     // H*HD

  const int*   x    = (const int*)  d_in[0];
  const float* emb  = (const float*)d_in[1];
  const float* wq   = (const float*)d_in[2];
  const float* bq   = (const float*)d_in[3];
  const float* wk   = (const float*)d_in[4];
  const float* bk   = (const float*)d_in[5];
  const float* wv   = (const float*)d_in[6];
  const float* bv   = (const float*)d_in[7];
  const float* wo   = (const float*)d_in[8];
  const float* bo   = (const float*)d_in[9];
  const float* ln1s = (const float*)d_in[10];
  const float* ln1b = (const float*)d_in[11];
  const float* w1   = (const float*)d_in[12];
  const float* b1   = (const float*)d_in[13];
  const float* w2   = (const float*)d_in[14];
  const float* b2   = (const float*)d_in[15];
  const float* ln2s = (const float*)d_in[16];
  const float* ln2b = (const float*)d_in[17];
  const float* ow   = (const float*)d_in[18];
  const float* ob   = (const float*)d_in[19];

  char* ws = (char*)d_ws;
  size_t off = 0;
  auto alloc = [&](size_t bytes) -> void* {
    void* p = ws + off;
    off += (bytes + 255) & ~(size_t)255;
    return p;
  };
  float*    h     = (float*)alloc((size_t)BS * D * 4);
  float*    res   = (float*)alloc((size_t)BS * D * 4);
  _Float16* hh16  = (_Float16*)alloc((size_t)BS * D * 2);
  _Float16* qh    = (_Float16*)alloc((size_t)BS * DH * 2);
  _Float16* kh    = (_Float16*)alloc((size_t)BS * DH * 2);
  _Float16* vh    = (_Float16*)alloc((size_t)BS * DH * 2);
  _Float16* aoh   = (_Float16*)alloc((size_t)BS * DH * 2);
  _Float16* ffh   = (_Float16*)alloc((size_t)BS * FF * 2);
  _Float16* wbufT = (_Float16*)alloc((size_t)D * OUT * 2);

  embed_pe_kernel<<<BS, 256, 0, stream>>>(x, emb, h, hh16);

  auto gemm = [&](const _Float16* A, const float* W, const float* bias,
                  float* C32, _Float16* C16, int K, int N, int act) {
    int wn = K * N;
    cvt_transpose_kernel<<<(wn + 255) / 256, 256, 0, stream>>>(W, wbufT, K, N);
    dim3 grid(N / GBN, BS / GBM);
    gemm_wmma_kernel<<<grid, 256, 0, stream>>>(A, wbufT, bias, C32, C16,
                                               BS, N, K, act);
  };

  for (int i = 0; i < NL; ++i) {
    gemm(hh16, wq + (size_t)i * D * DH, bq + i * DH, nullptr, qh, D, DH, 0);
    gemm(hh16, wk + (size_t)i * D * DH, bk + i * DH, nullptr, kh, D, DH, 0);
    gemm(hh16, wv + (size_t)i * D * DH, bv + i * DH, nullptr, vh, D, DH, 0);
    dim3 ag(B * Hh, S / 128);
    flash_attn_kernel<<<ag, 256, 0, stream>>>(qh, kh, vh, aoh);
    gemm(aoh, wo + (size_t)i * DH * D, bo + i * D, res, nullptr, DH, D, 0);
    residual_ln_kernel<<<BS, 256, 0, stream>>>(h, res, ln1s + i * D,
                                               ln1b + i * D, h, hh16);
    gemm(hh16, w1 + (size_t)i * D * FF, b1 + i * FF, nullptr, ffh, D, FF, 1);
    gemm(ffh, w2 + (size_t)i * FF * D, b2 + i * D, res, nullptr, FF, D, 0);
    residual_ln_kernel<<<BS, 256, 0, stream>>>(h, res, ln2s + i * D,
                                               ln2b + i * D, h, hh16);
  }
  gemm(hh16, ow, ob, (float*)d_out, nullptr, D, OUT, 0);
}
